// EGNN_network_13683765805085
// MI455X (gfx1250) — compile-verified, matrix operand order checked by reading, not run on hardware
//
#include <hip/hip_runtime.h>

// ---------------------------------------------------------------------------
// EGNN forward for MI455X (gfx1250): bf16 WMMA GEMMs, fused edge MLP with
// gather + 2x v_wmma_f32_16x16x32_bf16 GEMM + gate + f32 atomic scatter.
// All activations bf16 (L2-resident), accumulation f32.
// Wave->tile mapping: each wave owns a 16-column strip (unique B fragments,
// no redundant weight fetch from L2) and iterates the 4 row strips whose A
// fragments come from LDS. A fragments are pre-loaded into distinct regs so
// the 8 ds_load_b128 can issue as one clause and overlap the 4 WMMAs.
// ---------------------------------------------------------------------------

typedef __bf16 bf16_t;
typedef __attribute__((ext_vector_type(16))) __bf16 v16bf;
typedef __attribute__((ext_vector_type(8)))  float  v8f;

#define HID       128
#define NNODES    50000
#define NEDGES    800000
#define NGRAPHS   2500
#define TILE      64

static_assert(NEDGES % TILE == 0, "edge kernel assumes no tail");
static_assert(sizeof(bf16_t) == 2, "bf16 storage");

// ---- helpers ---------------------------------------------------------------

__device__ __forceinline__ float siluf(float v)    { return v / (1.f + __expf(-v)); }
__device__ __forceinline__ float sigmoidf(float v) { return 1.f / (1.f + __expf(-v)); }

__device__ __forceinline__ v8f zero8() {
  v8f z;
#pragma unroll
  for (int i = 0; i < 8; ++i) z[i] = 0.f;
  return z;
}

// 16x32 bf16 fragment (A: 16 rows x 32 K; B: 16 cols x 32 K from [N][K] layout).
// ISA 16-bit A layout: lanes 0-15 hold K=0..7 & 16..23, lanes 16-31 hold
// K=8..15 & 24..31 -> per lane two contiguous 16-byte chunks.
__device__ __forceinline__ v16bf load_frag16(const bf16_t* p, int ld, int lane) {
  const int half = (lane >> 4) & 1;
  const bf16_t* row = p + (lane & 15) * ld + half * 8;
  union { v16bf v; uint4 q[2]; } f;
  f.q[0] = *(const uint4*)(row);
  f.q[1] = *(const uint4*)(row + 16);
  return f.v;
}

__device__ __forceinline__ v8f wmma_bf16(v16bf a, v16bf b, v8f c) {
  return __builtin_amdgcn_wmma_f32_16x16x32_bf16(
      /*neg_a=*/false, a, /*neg_b=*/false, b,
      /*c_mod=*/(short)0, c, /*reuse_a=*/false, /*reuse_b=*/false);
}

// ---- small utility kernels -------------------------------------------------

__global__ void zero_kernel(float* p, int n) {
  int i = blockIdx.x * blockDim.x + threadIdx.x;
  if (i < n) p[i] = 0.f;
}

// W[K][N] fp32 -> Wt[N][K] bf16 (fragment-friendly transposed layout)
__global__ void convert_wt_kernel(const float* __restrict__ W, bf16_t* __restrict__ Wt,
                                  int K, int N) {
  int idx = blockIdx.x * blockDim.x + threadIdx.x;
  if (idx < K * N) {
    int k = idx / N, n = idx % N;
    Wt[n * K + k] = (bf16_t)W[idx];
  }
}

__global__ void embed_kernel(const int* __restrict__ z, const float* __restrict__ emb,
                             bf16_t* __restrict__ xb, int nNodes) {
  int idx = blockIdx.x * blockDim.x + threadIdx.x;
  if (idx < nNodes * HID) {
    int node = idx >> 7, c = idx & (HID - 1);
    xb[idx] = (bf16_t)emb[z[node] * HID + c];
  }
}

__global__ void dist_kernel(const int* __restrict__ edge, const float* __restrict__ pos,
                            float* __restrict__ dist, int nE) {
  int e = blockIdx.x * blockDim.x + threadIdx.x;
  if (e < nE) {
    int r = edge[e], s = edge[nE + e];
    float dx = pos[r * 3 + 0] - pos[s * 3 + 0];
    float dy = pos[r * 3 + 1] - pos[s * 3 + 1];
    float dz = pos[r * 3 + 2] - pos[s * 3 + 2];
    dist[e] = dx * dx + dy * dy + dz * dz;
  }
}

// ---- fused edge MLP: gather -> GEMM1(K=256)+rank1(dist) -> GEMM2 -> gate ->
//      f32 atomic scatter into agg. 64 edges per block, 8 waves. ------------

__global__ void __launch_bounds__(256)
edge_kernel(const bf16_t* __restrict__ xb,
            const int* __restrict__ recv, const int* __restrict__ send,
            const float* __restrict__ dist,
            const bf16_t* __restrict__ W1t,   // [128][256] bf16, n-major
            const float*  __restrict__ b1,
            const float*  __restrict__ wdist, // eW1 row 256 (fp32, [128])
            const bf16_t* __restrict__ W2t,   // [128][128] bf16, n-major
            const float*  __restrict__ b2,
            const float*  __restrict__ iW, const float* __restrict__ ibias,
            float* __restrict__ agg) {
  __shared__ bf16_t sA[TILE][264];   // [xr || xs] bf16, padded stride
  __shared__ bf16_t sM[TILE][136];   // msg1 bf16
  __shared__ float  sRed[TILE][4];
  __shared__ int    sRecv[TILE];
  __shared__ float  sDist[TILE];

  const int t = threadIdx.x;
  const int lane = t & 31;
  const int wave = t >> 5;
  const int e0 = blockIdx.x * TILE;

  // warm WGP$ with this block's weight working set (global_prefetch_b8)
  __builtin_prefetch(W1t + t * 128);              // 128*256 elems / 256 thr
  __builtin_prefetch(W2t + t * 64);               // 128*128 elems / 256 thr

  // gather: 4 threads per edge, 64 bf16 (128B) each
  {
    const int e = t >> 2, p = t & 3;
    const int ge = e0 + e;                        // no tail (E % 64 == 0)
    const int nid = (p < 2) ? recv[ge] : send[ge];
    if (p == 0) { sRecv[e] = nid; sDist[e] = dist[ge]; }
    const uint4* src = (const uint4*)(xb + (size_t)nid * HID + (p & 1) * 64);
    uint4* dst = (uint4*)&sA[e][(p & 1) * 64 + (p >> 1) * HID];
#pragma unroll
    for (int i = 0; i < 8; ++i) dst[i] = src[i];
  }
  __syncthreads();

  const int cbase = wave * 16;       // each wave owns a unique 16-col strip
  const int half = (lane >> 4) & 1;
  const int cn = lane & 15;

  // GEMM1: msg1 = silu(A @ W1 + b1 + dist ⊗ wdist)
  {
    v8f acc[4] = {zero8(), zero8(), zero8(), zero8()};
#pragma unroll
    for (int kk = 0; kk < 2 * HID; kk += 32) {
      v16bf b  = load_frag16(W1t + (size_t)cbase * 256 + kk, 256, lane);
      v16bf a0 = load_frag16(&sA[0][kk],  264, lane);
      v16bf a1 = load_frag16(&sA[16][kk], 264, lane);
      v16bf a2 = load_frag16(&sA[32][kk], 264, lane);
      v16bf a3 = load_frag16(&sA[48][kk], 264, lane);
      acc[0] = wmma_bf16(a0, b, acc[0]);
      acc[1] = wmma_bf16(a1, b, acc[1]);
      acc[2] = wmma_bf16(a2, b, acc[2]);
      acc[3] = wmma_bf16(a3, b, acc[3]);
    }
    const int col = cbase + cn;
    const float bb = b1[col], wd = wdist[col];
#pragma unroll
    for (int r = 0; r < 4; ++r)
#pragma unroll
      for (int i = 0; i < 8; ++i) {
        const int row = r * 16 + i + half * 8;
        sM[row][col] = (bf16_t)siluf(acc[r][i] + bb + sDist[row] * wd);
      }
  }
  __syncthreads();

  // GEMM2: msg2 = silu(msg1 @ W2 + b2), stored f32 in reused sA space
  float* sMsg2 = (float*)&sA[0][0];    // 64*128*4 = 32KB <= sizeof(sA)
  {
    v8f acc[4] = {zero8(), zero8(), zero8(), zero8()};
#pragma unroll
    for (int kk = 0; kk < HID; kk += 32) {
      v16bf b  = load_frag16(W2t + (size_t)cbase * HID + kk, HID, lane);
      v16bf a0 = load_frag16(&sM[0][kk],  136, lane);
      v16bf a1 = load_frag16(&sM[16][kk], 136, lane);
      v16bf a2 = load_frag16(&sM[32][kk], 136, lane);
      v16bf a3 = load_frag16(&sM[48][kk], 136, lane);
      acc[0] = wmma_bf16(a0, b, acc[0]);
      acc[1] = wmma_bf16(a1, b, acc[1]);
      acc[2] = wmma_bf16(a2, b, acc[2]);
      acc[3] = wmma_bf16(a3, b, acc[3]);
    }
    const int col = cbase + cn;
    const float bb = b2[col];
#pragma unroll
    for (int r = 0; r < 4; ++r)
#pragma unroll
      for (int i = 0; i < 8; ++i) {
        const int row = r * 16 + i + half * 8;
        sMsg2[row * HID + col] = siluf(acc[r][i] + bb);
      }
  }
  __syncthreads();

  // gate = sigmoid(msg2 . iW + ib); scatter msg2*gate into agg[recv]
  {
    const int e = t >> 2, q = t & 3;
    float part = 0.f;
#pragma unroll
    for (int j = 0; j < 32; ++j)
      part += sMsg2[e * HID + q * 32 + j] * iW[q * 32 + j];
    sRed[e][q] = part;
    __syncthreads();
    const float g =
        sigmoidf(sRed[e][0] + sRed[e][1] + sRed[e][2] + sRed[e][3] + ibias[0]);
    float* dstAgg = agg + (size_t)sRecv[e] * HID + q * 32;
#pragma unroll
    for (int j = 0; j < 32; ++j)
      atomicAdd(dstAgg + j, sMsg2[e * HID + q * 32 + j] * g);
  }
}

// ---- node update: x' = silu([x, agg] @ nW1 + nb1) @ nW2 + nb2 --------------

__global__ void __launch_bounds__(256)
node_kernel(const bf16_t* __restrict__ xb, const float* __restrict__ agg,
            const bf16_t* __restrict__ W1t, const float* __restrict__ b1,
            const bf16_t* __restrict__ W2t, const float* __restrict__ b2,
            bf16_t* __restrict__ xout, int nRows) {
  __shared__ bf16_t sA[TILE][264];
  __shared__ bf16_t sH[TILE][136];
  const int t = threadIdx.x, lane = t & 31, wave = t >> 5;
  const int r0 = blockIdx.x * TILE;

  __builtin_prefetch(W1t + t * 128);
  __builtin_prefetch(W2t + t * 64);

  {
    const int r = t >> 2, p = t & 3;
    int gr = r0 + r; if (gr > nRows - 1) gr = nRows - 1;   // clamp (tail-safe)
    if (p < 2) {
      const uint4* src = (const uint4*)(xb + (size_t)gr * HID + p * 64);
      uint4* dst = (uint4*)&sA[r][p * 64];
#pragma unroll
      for (int i = 0; i < 8; ++i) dst[i] = src[i];
    } else {
      const float4* src = (const float4*)(agg + (size_t)gr * HID + (p - 2) * 64);
      bf16_t* dst = &sA[r][HID + (p - 2) * 64];
#pragma unroll
      for (int i = 0; i < 16; ++i) {
        float4 v = src[i];
        dst[4 * i + 0] = (bf16_t)v.x; dst[4 * i + 1] = (bf16_t)v.y;
        dst[4 * i + 2] = (bf16_t)v.z; dst[4 * i + 3] = (bf16_t)v.w;
      }
    }
  }
  __syncthreads();

  const int cbase = wave * 16;
  const int half = (lane >> 4) & 1, cn = lane & 15;

  {
    v8f acc[4] = {zero8(), zero8(), zero8(), zero8()};
#pragma unroll
    for (int kk = 0; kk < 2 * HID; kk += 32) {
      v16bf b  = load_frag16(W1t + (size_t)cbase * 256 + kk, 256, lane);
      v16bf a0 = load_frag16(&sA[0][kk],  264, lane);
      v16bf a1 = load_frag16(&sA[16][kk], 264, lane);
      v16bf a2 = load_frag16(&sA[32][kk], 264, lane);
      v16bf a3 = load_frag16(&sA[48][kk], 264, lane);
      acc[0] = wmma_bf16(a0, b, acc[0]);
      acc[1] = wmma_bf16(a1, b, acc[1]);
      acc[2] = wmma_bf16(a2, b, acc[2]);
      acc[3] = wmma_bf16(a3, b, acc[3]);
    }
    const int col = cbase + cn;
    const float bb = b1[col];
#pragma unroll
    for (int r = 0; r < 4; ++r)
#pragma unroll
      for (int i = 0; i < 8; ++i)
        sH[r * 16 + i + half * 8][col] = (bf16_t)siluf(acc[r][i] + bb);
  }
  __syncthreads();

  {
    v8f acc[4] = {zero8(), zero8(), zero8(), zero8()};
#pragma unroll
    for (int kk = 0; kk < HID; kk += 32) {
      v16bf b  = load_frag16(W2t + (size_t)cbase * HID + kk, HID, lane);
      v16bf a0 = load_frag16(&sH[0][kk],  136, lane);
      v16bf a1 = load_frag16(&sH[16][kk], 136, lane);
      v16bf a2 = load_frag16(&sH[32][kk], 136, lane);
      v16bf a3 = load_frag16(&sH[48][kk], 136, lane);
      acc[0] = wmma_bf16(a0, b, acc[0]);
      acc[1] = wmma_bf16(a1, b, acc[1]);
      acc[2] = wmma_bf16(a2, b, acc[2]);
      acc[3] = wmma_bf16(a3, b, acc[3]);
    }
    const int col = cbase + cn;
    const float bb = b2[col];
#pragma unroll
    for (int r = 0; r < 4; ++r)
#pragma unroll
      for (int i = 0; i < 8; ++i) {
        const int gr = r0 + r * 16 + i + half * 8;
        if (gr < nRows) xout[(size_t)gr * HID + col] = (bf16_t)(acc[r][i] + bb);
      }
  }
}

// ---- readout: y = silu(x @ o1W1 + b) @ o1W2 + b ; atomic-add into yg[batch]-

__global__ void __launch_bounds__(256)
readout_node_kernel(const bf16_t* __restrict__ xb, const int* __restrict__ batch,
                    const bf16_t* __restrict__ W1t, const float* __restrict__ b1,
                    const bf16_t* __restrict__ W2t, const float* __restrict__ b2,
                    float* __restrict__ yg, int nRows) {
  __shared__ bf16_t sA[TILE][136];
  __shared__ bf16_t sH[TILE][136];
  __shared__ int sB[TILE];
  const int t = threadIdx.x, lane = t & 31, wave = t >> 5;
  const int r0 = blockIdx.x * TILE;

  __builtin_prefetch(W1t + t * 64);
  __builtin_prefetch(W2t + t * 64);

  {
    const int r = t >> 2, p = t & 3;
    int gr = r0 + r; if (gr > nRows - 1) gr = nRows - 1;
    if (p == 0) sB[r] = batch[gr];
    const uint4* src = (const uint4*)(xb + (size_t)gr * HID + p * 32);
    uint4* dst = (uint4*)&sA[r][p * 32];
#pragma unroll
    for (int i = 0; i < 4; ++i) dst[i] = src[i];
  }
  __syncthreads();

  const int cbase = wave * 16;
  const int half = (lane >> 4) & 1, cn = lane & 15;

  {
    v8f acc[4] = {zero8(), zero8(), zero8(), zero8()};
#pragma unroll
    for (int kk = 0; kk < HID; kk += 32) {
      v16bf b  = load_frag16(W1t + (size_t)cbase * HID + kk, HID, lane);
      v16bf a0 = load_frag16(&sA[0][kk],  136, lane);
      v16bf a1 = load_frag16(&sA[16][kk], 136, lane);
      v16bf a2 = load_frag16(&sA[32][kk], 136, lane);
      v16bf a3 = load_frag16(&sA[48][kk], 136, lane);
      acc[0] = wmma_bf16(a0, b, acc[0]);
      acc[1] = wmma_bf16(a1, b, acc[1]);
      acc[2] = wmma_bf16(a2, b, acc[2]);
      acc[3] = wmma_bf16(a3, b, acc[3]);
    }
    const int col = cbase + cn;
    const float bb = b1[col];
#pragma unroll
    for (int r = 0; r < 4; ++r)
#pragma unroll
      for (int i = 0; i < 8; ++i)
        sH[r * 16 + i + half * 8][col] = (bf16_t)siluf(acc[r][i] + bb);
  }
  __syncthreads();

  {
    v8f acc[4] = {zero8(), zero8(), zero8(), zero8()};
#pragma unroll
    for (int kk = 0; kk < HID; kk += 32) {
      v16bf b  = load_frag16(W2t + (size_t)cbase * HID + kk, HID, lane);
      v16bf a0 = load_frag16(&sH[0][kk],  136, lane);
      v16bf a1 = load_frag16(&sH[16][kk], 136, lane);
      v16bf a2 = load_frag16(&sH[32][kk], 136, lane);
      v16bf a3 = load_frag16(&sH[48][kk], 136, lane);
      acc[0] = wmma_bf16(a0, b, acc[0]);
      acc[1] = wmma_bf16(a1, b, acc[1]);
      acc[2] = wmma_bf16(a2, b, acc[2]);
      acc[3] = wmma_bf16(a3, b, acc[3]);
    }
    const int col = cbase + cn;
    const float bb = b2[col];
#pragma unroll
    for (int r = 0; r < 4; ++r)
#pragma unroll
      for (int i = 0; i < 8; ++i) {
        const int row = r * 16 + i + half * 8;
        const int gr = r0 + row;
        if (gr < nRows)
          atomicAdd(&yg[(size_t)sB[row] * HID + col], acc[r][i] + bb);
      }
  }
}

// ---- tiny graph head: out[g] = silu(yg[g] @ o2W1 + b) . o2W2 + b -----------

__global__ void __launch_bounds__(128)
graph_head_kernel(const float* __restrict__ yg,
                  const float* __restrict__ W1, const float* __restrict__ b1,
                  const float* __restrict__ W2, const float* __restrict__ b2,
                  float* __restrict__ out) {
  __shared__ float sY[HID];
  __shared__ float sRed[HID];
  const int g = blockIdx.x, j = threadIdx.x;
  sY[j] = yg[g * HID + j];
  __syncthreads();
  float acc = 0.f;
#pragma unroll 4
  for (int k = 0; k < HID; ++k) acc += sY[k] * W1[k * HID + j];
  sRed[j] = siluf(acc + b1[j]) * W2[j];
  __syncthreads();
  for (int s = 64; s > 0; s >>= 1) {
    if (j < s) sRed[j] += sRed[j + s];
    __syncthreads();
  }
  if (j == 0) out[g] = sRed[0] + b2[0];
}

// ---------------------------------------------------------------------------
// host launcher
// ---------------------------------------------------------------------------

extern "C" void kernel_launch(void* const* d_in, const int* in_sizes, int n_in,
                              void* d_out, int out_size, void* d_ws, size_t ws_size,
                              hipStream_t stream) {
  // --- resolve input layout (handles dict-insertion and jax-sorted orders) --
  const int *z, *edge, *batch;
  const float* pos;
  int base;
  const bool sortedTop = (in_sizes[1] != NNODES * 3);
  if (!sortedTop) {        // z, pos, edge_index, batch, params...
    z = (const int*)d_in[0]; pos = (const float*)d_in[1];
    edge = (const int*)d_in[2]; batch = (const int*)d_in[3];
    base = 4;              // emb
  } else {                 // batch, edge_index, params..., pos, z
    batch = (const int*)d_in[0]; edge = (const int*)d_in[1];
    base = 2;
    pos = (const float*)d_in[n_in - 2]; z = (const int*)d_in[n_in - 1];
  }
  const bool sortedParams = (in_sizes[base + 2] == HID * HID); // eW2 vs eb1
  const float* emb = (const float*)d_in[base];
  const float *eW1[2], *eb1[2], *eW2[2], *eb2[2], *iW[2], *ib[2];
  const float *nW1[2], *nb1[2], *nW2[2], *nb2[2];
  for (int L = 0; L < 2; ++L) {
    const int s = base + 1 + 10 * L;
    if (sortedParams) {    // eW1,eW2,eb1,eb2,iW,ib,nW1,nW2,nb1,nb2
      eW1[L] = (const float*)d_in[s + 0]; eW2[L] = (const float*)d_in[s + 1];
      eb1[L] = (const float*)d_in[s + 2]; eb2[L] = (const float*)d_in[s + 3];
      iW[L]  = (const float*)d_in[s + 4]; ib[L]  = (const float*)d_in[s + 5];
      nW1[L] = (const float*)d_in[s + 6]; nW2[L] = (const float*)d_in[s + 7];
      nb1[L] = (const float*)d_in[s + 8]; nb2[L] = (const float*)d_in[s + 9];
    } else {               // eW1,eb1,eW2,eb2,iW,ib,nW1,nb1,nW2,nb2
      eW1[L] = (const float*)d_in[s + 0]; eb1[L] = (const float*)d_in[s + 1];
      eW2[L] = (const float*)d_in[s + 2]; eb2[L] = (const float*)d_in[s + 3];
      iW[L]  = (const float*)d_in[s + 4]; ib[L]  = (const float*)d_in[s + 5];
      nW1[L] = (const float*)d_in[s + 6]; nb1[L] = (const float*)d_in[s + 7];
      nW2[L] = (const float*)d_in[s + 8]; nb2[L] = (const float*)d_in[s + 9];
    }
  }
  const int tb = base + 21;
  const float *o1W1, *o1b1, *o1W2, *o1b2, *o2W1, *o2b1, *o2W2, *o2b2;
  if (sortedParams) {      // o1W1,o1W2,o1b1,o1b2,o2W1,o2W2,o2b1,o2b2
    o1W1 = (const float*)d_in[tb + 0]; o1W2 = (const float*)d_in[tb + 1];
    o1b1 = (const float*)d_in[tb + 2]; o1b2 = (const float*)d_in[tb + 3];
    o2W1 = (const float*)d_in[tb + 4]; o2W2 = (const float*)d_in[tb + 5];
    o2b1 = (const float*)d_in[tb + 6]; o2b2 = (const float*)d_in[tb + 7];
  } else {                 // o1W1,o1b1,o1W2,o1b2,o2W1,o2b1,o2W2,o2b2
    o1W1 = (const float*)d_in[tb + 0]; o1b1 = (const float*)d_in[tb + 1];
    o1W2 = (const float*)d_in[tb + 2]; o1b2 = (const float*)d_in[tb + 3];
    o2W1 = (const float*)d_in[tb + 4]; o2b1 = (const float*)d_in[tb + 5];
    o2W2 = (const float*)d_in[tb + 6]; o2b2 = (const float*)d_in[tb + 7];
  }

  // --- workspace carve (≈ 56.2 MB) -----------------------------------------
  bf16_t* xbA = (bf16_t*)d_ws;                       // 12.8 MB
  bf16_t* xbB = xbA + (size_t)NNODES * HID;          // 12.8 MB
  float* agg  = (float*)(xbB + (size_t)NNODES * HID);// 25.6 MB
  float* dist = agg + (size_t)NNODES * HID;          // 3.2 MB
  float* yg   = dist + NEDGES;                       // 1.28 MB
  bf16_t* w   = (bf16_t*)(yg + NGRAPHS * HID);
  bf16_t *eW1t[2], *eW2t[2], *nW1t[2], *nW2t[2];
  for (int L = 0; L < 2; ++L) {
    eW1t[L] = w; w += 256 * HID;
    eW2t[L] = w; w += HID * HID;
    nW1t[L] = w; w += 256 * HID;
    nW2t[L] = w; w += HID * HID;
  }
  bf16_t* o1W1t = w; w += HID * HID;
  bf16_t* o1W2t = w; w += HID * HID;

  // --- weight conversion / transposition (bf16, [N][K]) --------------------
  auto cvt = [&](const float* W, bf16_t* Wt, int K, int N) {
    int n = K * N;
    convert_wt_kernel<<<(n + 255) / 256, 256, 0, stream>>>(W, Wt, K, N);
  };
  for (int L = 0; L < 2; ++L) {
    cvt(eW1[L], eW1t[L], 256, HID);  // first 256 rows; row 256 stays fp32
    cvt(eW2[L], eW2t[L], HID, HID);
    cvt(nW1[L], nW1t[L], 256, HID);
    cvt(nW2[L], nW2t[L], HID, HID);
  }
  cvt(o1W1, o1W1t, HID, HID);
  cvt(o1W2, o1W2t, HID, HID);

  // --- precompute -----------------------------------------------------------
  embed_kernel<<<(NNODES * HID + 255) / 256, 256, 0, stream>>>(z, emb, xbA, NNODES);
  dist_kernel<<<(NEDGES + 255) / 256, 256, 0, stream>>>(edge, pos, dist, NEDGES);

  // --- message-passing layers ----------------------------------------------
  const bf16_t* xin = xbA;
  bf16_t* xout = xbB;
  const int nodeGrid = (NNODES + TILE - 1) / TILE;
  for (int L = 0; L < 2; ++L) {
    zero_kernel<<<(NNODES * HID + 255) / 256, 256, 0, stream>>>(agg, NNODES * HID);
    edge_kernel<<<NEDGES / TILE, 256, 0, stream>>>(
        xin, edge, edge + NEDGES, dist,
        eW1t[L], eb1[L], eW1[L] + 256 * HID,
        eW2t[L], eb2[L], iW[L], ib[L], agg);
    node_kernel<<<nodeGrid, 256, 0, stream>>>(
        xin, agg, nW1t[L], nb1[L], nW2t[L], nb2[L], xout, NNODES);
    const bf16_t* tmp = xin; xin = xout; xout = (bf16_t*)tmp;
  }

  // --- readout --------------------------------------------------------------
  zero_kernel<<<(NGRAPHS * HID + 255) / 256, 256, 0, stream>>>(yg, NGRAPHS * HID);
  readout_node_kernel<<<nodeGrid, 256, 0, stream>>>(
      xin, batch, o1W1t, o1b1, o1W2t, o1b2, yg, NNODES);
  graph_head_kernel<<<NGRAPHS, 128, 0, stream>>>(
      yg, o2W1, o2b1, o2W2, o2b2, (float*)d_out);
}
